// DIMP_52175262712126
// MI455X (gfx1250) — matrix operand order repaired and checked by reading, not compiled, not run on hardware
//
#include <hip/hip_runtime.h>
#include <hip/hip_bf16.h>

typedef __attribute__((ext_vector_type(16))) __bf16       v16bf;
typedef __attribute__((ext_vector_type(8)))  __bf16       v8bf;
typedef __attribute__((ext_vector_type(8)))  float        v8f;
typedef __attribute__((ext_vector_type(4)))  float        v4f;
typedef __attribute__((ext_vector_type(4)))  unsigned int v4u;
typedef __attribute__((ext_vector_type(8)))  int          v8i;
typedef __attribute__((ext_vector_type(4)))  int          v4i;

#define D_FEAT 128
#define LDS_STRIDE 136   // 128 + 4 DWORDs pad -> 68 DWORDs/row, conflict-free ds_load_b128

// ---------------- zero fill (float4 stores) ----------------
__global__ __launch_bounds__(256) void k_zero(float* __restrict__ p, long n4) {
    long i = (long)blockIdx.x * 256 + threadIdx.x;
    if (i < n4) {
        v4f z; z[0] = 0.f; z[1] = 0.f; z[2] = 0.f; z[3] = 0.f;
        ((v4f*)p)[i] = z;
    }
}

// ---------------- weight convert: W[k][n] f32 -> Wt[n][k] bf16 ----------------
__global__ __launch_bounds__(256) void k_cvt_w(const float* __restrict__ W,
                                               __bf16* __restrict__ Wt) {
    int i = blockIdx.x * 256 + threadIdx.x;      // 0 .. 16383
    int n = i & (D_FEAT - 1);
    int k = i >> 7;
    Wt[n * D_FEAT + k] = (__bf16)W[k * D_FEAT + n];
}

// ---------------- edge scatter-add: Agg[dst] += H[src] ----------------
// one wave per edge, lane handles 4 contiguous features (float4 gather, 4 atomics)
__global__ __launch_bounds__(256) void k_scatter(const int* __restrict__ srcI,
                                                 const int* __restrict__ dstI,
                                                 const float* __restrict__ H,
                                                 float* __restrict__ Agg, int nE) {
    int t = blockIdx.x * 256 + threadIdx.x;
    int e = t >> 5;
    if (e >= nE) return;
    int f = (t & 31) << 2;
    int s = srcI[e];
    int d = dstI[e];
    v4f v = *(const v4f*)(H + (size_t)s * D_FEAT + f);
    float* o = Agg + (size_t)d * D_FEAT + f;
    atomicAdd(o + 0, v[0]);
    atomicAdd(o + 1, v[1]);
    atomicAdd(o + 2, v[2]);
    atomicAdd(o + 3, v[3]);
}

// ---------------- WMMA GEMM: Out = relu(Agg @ W + b) ----------------
// block = 256 threads = 8 waves; wave w owns the 16x16 tile (rows blockIdx*16, cols w*16).
// Wt (bf16, [n][k]) is DMAed into LDS by the Tensor Data Mover with row padding to
// LDS_STRIDE; the A strip is converted to bf16 once per block into LDS.
__global__ __launch_bounds__(256) void k_gemm_relu(const float* __restrict__ A,
                                                   const __bf16* __restrict__ Wt,
                                                   const float* __restrict__ bias,
                                                   float* __restrict__ Out, int nRows) {
    __shared__ __bf16 shA[16 * LDS_STRIDE];        //  4.25 KB
    __shared__ __bf16 shW[D_FEAT * LDS_STRIDE];    // 34 KB (TDM destination)

    const int tid = threadIdx.x;

    // ---- TDM: DMA Wt[128][128] bf16 -> shW with 4-DWORD row padding ----
    if (tid == 0) {
        unsigned long long ga = (unsigned long long)(size_t)Wt;
        unsigned int ldsOff   = (unsigned int)(size_t)(&shW[0]);
        v4u g0;
        g0[0] = 1u;                                            // count=1 (valid), user mode
        g0[1] = ldsOff;                                        // lds_addr
        g0[2] = (unsigned int)(ga & 0xFFFFFFFFull);            // global_addr[31:0]
        g0[3] = (unsigned int)((ga >> 32) & 0x1FFFFFFull)      // global_addr[56:32]
              | (2u << 30);                                    // type = 2 ("image")
        v8i g1;
        g1[0] = (int)((1u << 16)      // data_size = 1 -> 2 bytes
              |       (1u << 20)      // pad_enable
              |       (5u << 22)      // pad_interval: every 64 DWORDs (256B = one row)
              |       (3u << 25));    // pad_amount: 4 DWORDs -> 272B row stride
        g1[1] = (int)(128u << 16);    // tensor_dim0[15:0] = 128 (bits 79:48 start here)
        g1[2] = (int)(128u << 16);    // tensor_dim0 hi = 0 ; tensor_dim1[15:0] = 128
        g1[3] = (int)(128u << 16);    // tensor_dim1 hi = 0 ; tile_dim0 = 128
        g1[4] = 128;                  // tile_dim1 = 128 ; tile_dim2 = 0
        g1[5] = 128;                  // tensor_dim0_stride = 128 elements
        g1[6] = 0;                    // stride hi / tensor_dim1_stride (unused, 2D)
        g1[7] = 0;
        v4i z4; z4[0] = 0; z4[1] = 0; z4[2] = 0; z4[3] = 0;
        v8i z8; z8[0] = 0; z8[1] = 0; z8[2] = 0; z8[3] = 0;
        z8[4] = 0; z8[5] = 0; z8[6] = 0; z8[7] = 0;
        __builtin_amdgcn_tensor_load_to_lds(g0, g1, z4, z4, z8, 0);
    }

    // ---- stage A strip: 16 rows x 128 f32 -> bf16 in LDS (once per block) ----
    {
        int r  = tid >> 4;            // 0..15
        int c0 = (tid & 15) * 8;      // 0,8,...,120
        int ar = blockIdx.x * 16 + r;
        if (ar >= nRows) ar = nRows - 1;
        const float* ap = A + (size_t)ar * D_FEAT + c0;
        v4f x0 = *(const v4f*)(ap);
        v4f x1 = *(const v4f*)(ap + 4);
        v8bf pk;
#pragma unroll
        for (int i = 0; i < 4; ++i) {
            pk[i]     = (__bf16)x0[i];
            pk[i + 4] = (__bf16)x1[i];
        }
        *(v8bf*)(shA + r * LDS_STRIDE + c0) = pk;
    }

    __builtin_amdgcn_s_wait_tensorcnt(0);
    __syncthreads();

    // ---- WMMA main loop: fragments from LDS ----
    int wave = tid >> 5;              // column tile
    int lane = tid & 31;
    int row  = lane & 15;             // A row / B,C,D column within tile
    int hi   = lane >> 4;
    int klo  = hi * 8;                // A-fragment K sub-offset
    int hi16 = hi * 16;               // unused; B fragment uses klo

    const __bf16* aBase = shA + row * LDS_STRIDE;
    const __bf16* bBase = shW + (wave * 16 + row) * LDS_STRIDE;

    v8f c = {};
#pragma unroll
    for (int k0 = 0; k0 < D_FEAT; k0 += 32) {
        // A fragment: lane holds K = {k0+klo .. +7} and {k0+klo+16 .. +23}
        v8bf al = *(const v8bf*)(aBase + k0 + klo);
        v8bf ah = *(const v8bf*)(aBase + k0 + klo + 16);
        v16bf a = __builtin_shufflevector(al, ah, 0, 1, 2, 3, 4, 5, 6, 7,
                                                  8, 9, 10, 11, 12, 13, 14, 15);
        // B fragment: lane = column, K = {k0+klo .. +7} and {k0+klo+16 .. +23}
        v8bf bl = *(const v8bf*)(bBase + k0 + klo);
        v8bf bh = *(const v8bf*)(bBase + k0 + klo + 16);
        v16bf b = __builtin_shufflevector(bl, bh, 0, 1, 2, 3, 4, 5, 6, 7,
                                                  8, 9, 10, 11, 12, 13, 14, 15);
        c = __builtin_amdgcn_wmma_f32_16x16x32_bf16(
                /*neg_a=*/false, a, /*neg_b=*/false, b,
                /*c_mod=*/(short)0, c, /*reuse_a=*/false, /*reuse_b=*/false);
    }

    // ---- epilogue: bias + relu + store ----
    // D layout: lane holds column (lane&15), rows hi*8 + r for VGPR r
    int tileM = blockIdx.x * 16;
    int coln  = wave * 16 + row;
    float bv  = bias[coln];
    if (tileM + 16 <= nRows) {
        // full tile (always, since grid is exact for nRows % 16 == 0): no per-row guards
        float* obase = Out + (size_t)(tileM + hi * 8) * D_FEAT + coln;
#pragma unroll
        for (int r = 0; r < 8; ++r) {
            float v = c[r] + bv;
            obase[(size_t)r * D_FEAT] = v > 0.f ? v : 0.f;
        }
    } else {
        // ragged tail tile
#pragma unroll
        for (int r = 0; r < 8; ++r) {
            int m = tileM + hi * 8 + r;
            if (m < nRows) {
                float v = c[r] + bv;
                Out[(size_t)m * D_FEAT + coln] = v > 0.f ? v : 0.f;
            }
        }
    }
}

// ---------------- column sums (for graph summary mean) ----------------
__global__ __launch_bounds__(256) void k_colsum(const float* __restrict__ H,
                                                float* __restrict__ S, int nRows) {
    int col  = threadIdx.x & 127;
    int half = threadIdx.x >> 7;          // 0 or 1
    int r0   = blockIdx.x * 128 + half;
    int r1   = blockIdx.x * 128 + 128;
    if (r1 > nRows) r1 = nRows;
    float acc = 0.f;
    for (int r = r0; r < r1; r += 2) acc += H[(size_t)r * D_FEAT + col];
    atomicAdd(&S[col], acc);
}

// ---------------- wsv = Wd @ (S / N) ----------------
__global__ __launch_bounds__(128) void k_matvec(const float* __restrict__ Wd,
                                                const float* __restrict__ S,
                                                float* __restrict__ WSV, float invN) {
    __shared__ float ssh[D_FEAT];
    int t = threadIdx.x;
    ssh[t] = S[t] * invN;
    __syncthreads();
    float acc = 0.f;
#pragma unroll 8
    for (int k = 0; k < D_FEAT; ++k) acc += Wd[t * D_FEAT + k] * ssh[k];
    WSV[t] = acc;
}

// ---------------- per-node dot: out[i] = H[i] . wsv (wave per node) ----------------
__global__ __launch_bounds__(256) void k_dots(const float* __restrict__ H,
                                              const float* __restrict__ WSV,
                                              float* __restrict__ out, int n) {
    int wave = threadIdx.x >> 5;
    int lane = threadIdx.x & 31;
    int node = blockIdx.x * 8 + wave;
    if (node >= n) return;
    v4f h4 = *(const v4f*)(H + (size_t)node * D_FEAT + lane * 4);
    v4f w4 = *(const v4f*)(WSV + lane * 4);
    float p = h4[0] * w4[0] + h4[1] * w4[1] + h4[2] * w4[2] + h4[3] * w4[3];
#pragma unroll
    for (int off = 16; off >= 1; off >>= 1) p += __shfl_down(p, off);
    if (lane == 0) out[node] = p;
}

extern "C" void kernel_launch(void* const* d_in, const int* in_sizes, int n_in,
                              void* d_out, int out_size, void* d_ws, size_t ws_size,
                              hipStream_t stream) {
    const int*   edge = (const int*)d_in[0];
    const float* x    = (const float*)d_in[1];
    const float* xs   = (const float*)d_in[2];
    const float* W[3] = {(const float*)d_in[3], (const float*)d_in[5], (const float*)d_in[7]};
    const float* B[3] = {(const float*)d_in[4], (const float*)d_in[6], (const float*)d_in[8]};
    const float* Wd   = (const float*)d_in[9];

    const int nE = in_sizes[0] / 2;
    const int nN = in_sizes[1] / D_FEAT;
    const int* srcI = edge;
    const int* dstI = edge + nE;
    float* out = (float*)d_out;

    // ---- workspace carve-up (256B aligned) ----
    auto alignup = [](size_t v) { return (v + 255) & ~(size_t)255; };
    char* ws = (char*)d_ws;
    size_t featB = alignup((size_t)nN * D_FEAT * sizeof(float));
    float*  agg   = (float*)ws;                 ws += featB;
    float*  hA    = (float*)ws;                 ws += featB;
    float*  hB    = (float*)ws;                 ws += featB;
    __bf16* Wt[3];
    for (int l = 0; l < 3; ++l) { Wt[l] = (__bf16*)ws; ws += alignup(D_FEAT * D_FEAT * sizeof(__bf16)); }
    float*  Ssum  = (float*)ws;                 ws += alignup(D_FEAT * sizeof(float));
    float*  WSV   = (float*)ws;                 ws += alignup(D_FEAT * sizeof(float));

    // ---- one-time (per launch) weight transpose+convert ----
    for (int l = 0; l < 3; ++l)
        k_cvt_w<<<(D_FEAT * D_FEAT) / 256, 256, 0, stream>>>(W[l], Wt[l]);

    const long n4       = (long)nN * D_FEAT / 4;
    const int  zeroGrid = (int)((n4 + 255) / 256);
    const int  scatGrid = (nE * 32 + 255) / 256;
    const int  gemmGrid = (nN + 15) / 16;

    // ---- two encode passes ----
    for (int pass = 0; pass < 2; ++pass) {
        const float* cur = pass ? xs : x;
        float* h = pass ? hB : hA;
        for (int l = 0; l < 3; ++l) {
            k_zero<<<zeroGrid, 256, 0, stream>>>(agg, n4);
            k_scatter<<<scatGrid, 256, 0, stream>>>(srcI, dstI, cur, agg, nE);
            k_gemm_relu<<<gemmGrid, 256, 0, stream>>>(agg, Wt[l], B[l], h, nN);
            cur = h;
        }
    }

    // ---- summary + discriminator ----
    k_zero<<<1, 256, 0, stream>>>(Ssum, D_FEAT / 4);
    k_colsum<<<(nN + 127) / 128, 256, 0, stream>>>(hA, Ssum, nN);
    k_matvec<<<1, 128, 0, stream>>>(Wd, Ssum, WSV, 1.0f / (float)nN);
    const int dotGrid = (nN + 7) / 8;
    k_dots<<<dotGrid, 256, 0, stream>>>(hA, WSV, out, nN);
    k_dots<<<dotGrid, 256, 0, stream>>>(hB, WSV, out + nN, nN);
}
